// GLABlock_60859686584552
// MI455X (gfx1250) — compile-verified, matrix operand order checked by reading, not run on hardware
//
#include <hip/hip_runtime.h>
#include <hip/hip_bf16.h>

// GLA block forward for MI455X (gfx1250, wave32, WMMA + TDM).
// Pipeline: cvt(bf16) -> GEMM qkvg -> lowrank gk -> chunked scan (WMMA) ->
//           rmsnorm+silu -> GEMM out.
// Workspace use: ~720 MB in d_ws.

typedef __attribute__((ext_vector_type(16))) __bf16 v16bf;
typedef __attribute__((ext_vector_type(8)))  float  v8f;
typedef unsigned int u32x4 __attribute__((ext_vector_type(4)));
typedef int          i32x8 __attribute__((ext_vector_type(8)));
typedef int          i32x4 __attribute__((ext_vector_type(4)));

#define DEV static __device__ __forceinline__

#if defined(__has_builtin)
#  if __has_builtin(__builtin_amdgcn_tensor_load_to_lds) && \
      __has_builtin(__builtin_amdgcn_s_wait_tensorcnt)
#    define HAVE_TDM 1
#  endif
#endif
#ifndef HAVE_TDM
#  define HAVE_TDM 0
#endif

constexpr int   T_LEN = 4096;
constexpr int   D_IN  = 1024;
constexpr int   DK    = 512;
constexpr int   DV    = 1024;
constexpr int   NCAT  = DK + DK + DV + DV;   // 3072
constexpr long  MROWS = 8L * T_LEN;          // 32768

// ---- WMMA fragment helpers (layouts per CDNA5 ISA 7.12.2) -----------------

DEV int kmap(int lane, int e) {
  int r = e >> 1, p = e & 1;
  return ((lane & 16) ? 8 : 0) + ((r & 4) ? 16 : 0) + ((r & 3) << 1) + p;
}

// base points at element [row_or_col + (lane&15)][k0] of a K-contiguous tile
DEV v16bf frag_ld(const __bf16* base, int lane) {
  v16bf f;
#pragma unroll
  for (int e = 0; e < 16; ++e) f[e] = base[kmap(lane, e)];
  return f;
}

DEV v8f wmma_bf16(v16bf a, v16bf b, v8f c) {
  return __builtin_amdgcn_wmma_f32_16x16x32_bf16(false, a, false, b,
                                                 (short)0, c, false, false);
}

// ---- TDM descriptor issue: load a [128 x 32] bf16 tile from row-major -----
// [MROWS x K] tensor at (row0, k0) into packed LDS (ISA 08_async_tensor §8).

DEV void tdm_issue_a(const __bf16* A, long row0, int k0, int K,
                     unsigned lds_off) {
#if HAVE_TDM
  unsigned long long ga = (unsigned long long)(size_t)&A[row0 * (long)K + k0];
  unsigned Ku = (unsigned)K, Mu = (unsigned)MROWS;
  u32x4 g0;
  g0[0] = 1u;                                   // count=1, no gather
  g0[1] = lds_off;                              // lds_addr (bytes)
  g0[2] = (unsigned)ga;                         // global_addr[95:64]
  g0[3] = (unsigned)(ga >> 32) | (2u << 30);    // addr msbs | type=2 (image)
  i32x8 g1;
  g1[0] = (int)(1u << 16);                      // data_size=1 (2 bytes)
  g1[1] = (int)((Ku & 0xffffu) << 16);          // tensor_dim0[15:0]
  g1[2] = (int)((Ku >> 16) | ((Mu & 0xffffu) << 16));   // td0 hi | td1 lo
  g1[3] = (int)((Mu >> 16) | (32u << 16));      // td1 hi | tile_dim0=32
  g1[4] = 128;                                  // tile_dim1=128
  g1[5] = (int)Ku;                              // tensor_dim0_stride[31:0]
  g1[6] = 0;
  g1[7] = 0;
  i32x4 z4 = {};
#if defined(__clang_major__) && __clang_major__ >= 23
  i32x8 z8 = {};
  __builtin_amdgcn_tensor_load_to_lds(g0, g1, z4, z4, z8, 0);
#else
  __builtin_amdgcn_tensor_load_to_lds(g0, g1, z4, z4, 0);
#endif
#endif
}

// ---- manual tile staging (each helper: 2 global loads before its LDS stores) --------

DEV void stage_a(const __bf16* __restrict__ A, long row0, int K, int k0,
                 int t, __bf16 (*dA)[32]) {
  int c0 = t, c1 = t + 256;
  int r0 = c0 >> 2, ko0 = (c0 & 3) * 8;
  int r1 = c1 >> 2, ko1 = (c1 & 3) * 8;
  uint4 a0 = *(const uint4*)&A[(row0 + r0) * K + k0 + ko0];
  uint4 a1 = *(const uint4*)&A[(row0 + r1) * K + k0 + ko1];
  *(uint4*)&dA[r0][ko0] = a0;
  *(uint4*)&dA[r1][ko1] = a1;
}

DEV void stage_b(const __bf16* __restrict__ B, long col0, int N, int k0,
                 int t, __bf16 (*dB)[32]) {
  int c0 = t, c1 = t + 256;
  int kk0 = c0 >> 4, no0 = (c0 & 15) * 8;
  int kk1 = c1 >> 4, no1 = (c1 & 15) * 8;
  uint4 b0 = *(const uint4*)&B[(long)(k0 + kk0) * N + col0 + no0];
  uint4 b1 = *(const uint4*)&B[(long)(k0 + kk1) * N + col0 + no1];
  __bf16 tb[8];
  *(uint4*)tb = b0;
#pragma unroll
  for (int e = 0; e < 8; ++e) dB[no0 + e][kk0] = tb[e];
  *(uint4*)tb = b1;
#pragma unroll
  for (int e = 0; e < 8; ++e) dB[no1 + e][kk1] = tb[e];
}

// ---- conversion kernels ----------------------------------------------------

__global__ void k_cvt(const float* __restrict__ s, __bf16* __restrict__ d) {
  long i = (long)blockIdx.x * 256 + threadIdx.x;
  d[i] = (__bf16)s[i];
}

__global__ void k_wcat(const float* __restrict__ Wq, const float* __restrict__ Wk,
                       const float* __restrict__ Wv, const float* __restrict__ Wg,
                       __bf16* __restrict__ Wcat) {
  long idx = (long)blockIdx.x * 256 + threadIdx.x;   // over 1024*3072
  int k = (int)(idx / NCAT), c = (int)(idx % NCAT);
  float v;
  if      (c < 512)  v = Wq[k * 512  + c];
  else if (c < 1024) v = Wk[k * 512  + (c - 512)];
  else if (c < 2048) v = Wv[k * 1024 + (c - 1024)];
  else               v = Wg[k * 1024 + (c - 2048)];
  Wcat[idx] = (__bf16)v;
}

// ---- generic 128x128x(BK=32) bf16 GEMM, f32 out, double-buffered ----------
// grid = (N/128, M/128), block = 256 (8 waves). Wave: 32x64 region.
// A tile staged by TDM (tensor_load_to_lds) when available, else VALU path.

__global__ __launch_bounds__(256)
void k_gemm_bf16(const __bf16* __restrict__ A, const __bf16* __restrict__ B,
                 float* __restrict__ C, int K, int N) {
  __shared__ __bf16 sA[2][128][32];   // [m][k]
  __shared__ __bf16 sB[2][128][32];   // [n][k]  (K-contiguous per column)
  const int t = threadIdx.x, lane = t & 31, w = t >> 5;
  const long row0 = (long)blockIdx.y * 128;
  const long col0 = (long)blockIdx.x * 128;
  const int wr = w >> 1, wc = w & 1;
  const bool issuer = (t == 0);
  v8f acc[2][4] = {};

  // prologue: stage tile 0 into buffer 0
#if HAVE_TDM
  if (issuer) tdm_issue_a(A, row0, 0, K, (unsigned)(size_t)&sA[0][0][0]);
#else
  stage_a(A, row0, K, 0, t, sA[0]);
#endif
  stage_b(B, col0, N, 0, t, sB[0]);
#if HAVE_TDM
  if (issuer) __builtin_amdgcn_s_wait_tensorcnt(0);
#endif
  __syncthreads();

  for (int k0 = 0; k0 < K; k0 += 32) {
    const int p = (k0 >> 5) & 1;
    const bool more = (k0 + 32) < K;
    if (more) {                           // stage next tile into 1-p
#if HAVE_TDM
      if (issuer)
        tdm_issue_a(A, row0, k0 + 32, K, (unsigned)(size_t)&sA[1 - p][0][0]);
#else
      stage_a(A, row0, K, k0 + 32, t, sA[1 - p]);
#endif
      stage_b(B, col0, N, k0 + 32, t, sB[1 - p]);
      __builtin_prefetch(&B[(long)(k0 + 64 + (t >> 4)) * N + col0], 0, 1);
    }
    v16bf af[2], bfr[4];
#pragma unroll
    for (int i = 0; i < 2; ++i)
      af[i] = frag_ld(&sA[p][wr * 32 + i * 16 + (lane & 15)][0], lane);
#pragma unroll
    for (int j = 0; j < 4; ++j)
      bfr[j] = frag_ld(&sB[p][wc * 64 + j * 16 + (lane & 15)][0], lane);
#pragma unroll
    for (int i = 0; i < 2; ++i)
#pragma unroll
      for (int j = 0; j < 4; ++j) acc[i][j] = wmma_bf16(af[i], bfr[j], acc[i][j]);
#if HAVE_TDM
    if (more && issuer) __builtin_amdgcn_s_wait_tensorcnt(0);
#endif
    __syncthreads();
  }

  const int hi8 = (lane & 16) ? 8 : 0, cl = lane & 15;
#pragma unroll
  for (int i = 0; i < 2; ++i)
#pragma unroll
    for (int j = 0; j < 4; ++j) {
      long rb = row0 + wr * 32 + i * 16 + hi8;
      long cb = col0 + wc * 64 + j * 16 + cl;
#pragma unroll
      for (int r = 0; r < 8; ++r) C[(rb + r) * (long)N + cb] = acc[i][j][r];
    }
}

// ---- low-rank path: h = x @ Wgk1 (N=16) -----------------------------------
// grid = M/128, block = 256; wave w handles rows [16w,16w+16).

__global__ __launch_bounds__(256)
void k_gemm_h(const __bf16* __restrict__ xb, const __bf16* __restrict__ w1b,
              float* __restrict__ h) {
  __shared__ __bf16 sA[128][32];
  __shared__ __bf16 sB[16][32];
  const int t = threadIdx.x, lane = t & 31, w = t >> 5;
  const long row0 = (long)blockIdx.x * 128;
  v8f acc = {};
  for (int k0 = 0; k0 < D_IN; k0 += 32) {
    stage_a(xb, row0, D_IN, k0, t, sA);
#pragma unroll
    for (int e = 0; e < 2; ++e) {                 // 512 elems transposed
      int id = t * 2 + e;
      int kk = id >> 4, n = id & 15;
      sB[n][kk] = w1b[(k0 + kk) * 16 + n];
    }
    __syncthreads();
    v16bf af = frag_ld(&sA[w * 16 + (lane & 15)][0], lane);
    v16bf bf = frag_ld(&sB[lane & 15][0], lane);
    acc = wmma_bf16(af, bf, acc);
    __syncthreads();
  }
  const int hi8 = (lane & 16) ? 8 : 0, cl = lane & 15;
#pragma unroll
  for (int r = 0; r < 8; ++r) h[(row0 + w * 16 + hi8 + r) * 16 + cl] = acc[r];
}

// ---- gk = logsigmoid(h @ Wgk2 + b) / 16  (K=16, memory bound) -------------

__global__ void k_gk_act(const float* __restrict__ h, const float* __restrict__ W2,
                         const float* __restrict__ b2, float* __restrict__ gk) {
  long idx = (long)blockIdx.x * 256 + threadIdx.x;  // over M*512
  int m = (int)(idx >> 9), n = (int)(idx & 511);
  float z = b2[n];
#pragma unroll
  for (int r = 0; r < 16; ++r) z += h[m * 16 + r] * W2[r * 512 + n];
  float ls = fminf(z, 0.f) - log1pf(__expf(-fabsf(z)));
  gk[idx] = ls * (1.0f / 16.0f);
}

// ---- chunked GLA scan ------------------------------------------------------
// grid = B*H*2 = 128 blocks, block = 256 (8 waves).
// Each block: one (batch, head, dv-half). S (64x64 f32) lives in wave VGPRs.

__global__ __launch_bounds__(256)
void k_attn(const float* __restrict__ proj, const float* __restrict__ gkb,
            float* __restrict__ obuf) {
  __shared__ float  sB[64][64];      // cumsum gates b (f32); first 8KB aliased as sAm
  __shared__ float  sBlast[64];
  __shared__ __bf16 sQe[64][64];     // [i][kk]
  __shared__ __bf16 sKe[64][64];     // [j][kk]
  __shared__ __bf16 sKdT[64][64];    // [kk][i]
  __shared__ __bf16 sVT[64][64];     // [vv][i]
  __shared__ __bf16 sSb[64][64];     // [vv][kk] bf16 copy of state
  __bf16* sAm = (__bf16*)sB;         // [i][j], alias (phase-separated)

  const int t = threadIdx.x, lane = t & 31, w = t >> 5;
  const int hi8 = (lane & 16) ? 8 : 0, cl = lane & 15;
  const int vblk = blockIdx.x & 1;
  const int hh   = (blockIdx.x >> 1) & 7;
  const int bat  = blockIdx.x >> 4;
  const int qoff = hh * 64, koff = 512 + hh * 64;
  const int voff = 1024 + hh * 128 + vblk * 64;
  const int gkoff = hh * 64;
  const int ri = t >> 2, cb = (t & 3) * 16;

  v8f Sreg[2] = {};                  // persistent state fragments (this wave's 2 tiles)
#pragma unroll
  for (int j = 0; j < 16; ++j) ((__bf16*)sSb)[t + 256 * j] = (__bf16)0.f;

  for (int c = 0; c < 64; ++c) {
    const long mbase = (long)bat * T_LEN + (long)c * 64;
    { // (a) raw gk chunk -> sB
      const float* src = &gkb[(mbase + ri) * 512 + gkoff + cb];
#pragma unroll
      for (int jj = 0; jj < 4; ++jj)
        *(float4*)&sB[ri][cb + jj * 4] = *(const float4*)&src[jj * 4];
    }
    __syncthreads();
    if (t < 64) {                    // (b) cumsum over chunk position, per kk
      float run = 0.f;
      for (int i = 0; i < 64; ++i) { run += sB[i][t]; sB[i][t] = run; }
      sBlast[t] = run;
    }
    __syncthreads();
    { // (c,d) gate q/k, build kd^T and v^T tiles (f32 math, bf16 stores)
      const float* qs = &proj[(mbase + ri) * NCAT + qoff + cb];
      const float* ks = &proj[(mbase + ri) * NCAT + koff + cb];
      const float* vs = &proj[(mbase + ri) * NCAT + voff + cb];
#pragma unroll 4
      for (int jj = 0; jj < 16; ++jj) {
        int kk = cb + jj;
        float bc = sB[ri][kk];
        float qv = qs[jj], kv = ks[jj];
        sQe[ri][kk]  = (__bf16)(qv * __expf(bc) * 0.125f);   // scale = dk^-0.5
        sKe[ri][kk]  = (__bf16)(kv * __expf(-bc));
        sKdT[kk][ri] = (__bf16)(kv * __expf(sBlast[kk] - bc));
        sVT[kk][ri]  = (__bf16)vs[jj];                       // kk == vv index here
      }
    }
    __syncthreads();
    // phase 2: A = qe @ ke^T, causal mask, -> sAm (bf16)
#pragma unroll
    for (int s = 0; s < 2; ++s) {
      int tid = w * 2 + s, tm = tid >> 2, tn = tid & 3;
      v8f acc = {};
#pragma unroll
      for (int kb = 0; kb < 64; kb += 32)
        acc = wmma_bf16(frag_ld(&sQe[tm * 16 + cl][kb], lane),
                        frag_ld(&sKe[tn * 16 + cl][kb], lane), acc);
      int j = tn * 16 + cl;
#pragma unroll
      for (int r = 0; r < 8; ++r) {
        int i = tm * 16 + hi8 + r;
        sAm[i * 64 + j] = (__bf16)((j <= i) ? acc[r] : 0.f);
      }
    }
    __syncthreads();
    // phase 3a: O = A@V + qe@S  -> global (f32)
#pragma unroll
    for (int s = 0; s < 2; ++s) {
      int tid = w * 2 + s, tm = tid >> 2, tn = tid & 3;
      v8f acc = {};
#pragma unroll
      for (int kb = 0; kb < 64; kb += 32) {
        acc = wmma_bf16(frag_ld(&sAm[(tm * 16 + cl) * 64 + kb], lane),
                        frag_ld(&sVT[tn * 16 + cl][kb], lane), acc);
        acc = wmma_bf16(frag_ld(&sQe[tm * 16 + cl][kb], lane),
                        frag_ld(&sSb[tn * 16 + cl][kb], lane), acc);
      }
      long row = mbase + tm * 16 + hi8;
      int  col = hh * 128 + vblk * 64 + tn * 16 + cl;
#pragma unroll
      for (int r = 0; r < 8; ++r) obuf[(row + r) * (long)DV + col] = acc[r];
    }
    __syncthreads();                 // all sSb reads done before rewrite
    // phase 3b: dS = kd^T @ V ; S = S*exp(b_last) + dS ; publish bf16 S
#pragma unroll
    for (int s = 0; s < 2; ++s) {
      int tid = w * 2 + s, tm = tid >> 2, tn = tid & 3;   // tm: kk-block, tn: vv-block
      v8f acc = {};
#pragma unroll
      for (int kb = 0; kb < 64; kb += 32)
        acc = wmma_bf16(frag_ld(&sKdT[tm * 16 + cl][kb], lane),
                        frag_ld(&sVT[tn * 16 + cl][kb], lane), acc);
      int vv = tn * 16 + cl;
#pragma unroll
      for (int r = 0; r < 8; ++r) {
        int kk = tm * 16 + hi8 + r;
        float ns = Sreg[s][r] * __expf(sBlast[kk]) + acc[r];
        Sreg[s][r] = ns;
        sSb[vv][kk] = (__bf16)ns;
      }
    }
    __syncthreads();
  }
}

// ---- RMS-norm over dv=128 per head + SiLU gate -> bf16 --------------------
// grid = M rows, block = 256; wave w == head w (32 lanes x 4 elems = 128).

__global__ __launch_bounds__(256)
void k_norm(const float* __restrict__ obuf, const float* __restrict__ proj,
            const float* __restrict__ nw, __bf16* __restrict__ on) {
  long m = blockIdx.x;
  int t = threadIdx.x, lane = t & 31, hh = t >> 5;
  int j0 = lane * 4;
  float4 ov = *(const float4*)&obuf[m * DV + hh * 128 + j0];
  float ss = ov.x * ov.x + ov.y * ov.y + ov.z * ov.z + ov.w * ov.w;
#pragma unroll
  for (int off = 16; off > 0; off >>= 1) ss += __shfl_xor(ss, off, 32);
  float r = rsqrtf(ss * (1.f / 128.f) + 1e-5f);
  float4 gv = *(const float4*)&proj[m * NCAT + 2048 + hh * 128 + j0];
  float4 nv = *(const float4*)&nw[j0];
  float vals[4] = {ov.x, ov.y, ov.z, ov.w};
  float gs[4]   = {gv.x, gv.y, gv.z, gv.w};
  float ns[4]   = {nv.x, nv.y, nv.z, nv.w};
  __bf16 outv[4];
#pragma unroll
  for (int e = 0; e < 4; ++e) {
    float g = gs[e];
    float silu = g / (1.f + __expf(-g));
    outv[e] = (__bf16)(vals[e] * r * ns[e] * silu);
  }
  *(uint2*)&on[m * DV + hh * 128 + j0] = *(uint2*)outv;
}

// ---- host ------------------------------------------------------------------

extern "C" void kernel_launch(void* const* d_in, const int* in_sizes, int n_in,
                              void* d_out, int out_size, void* d_ws, size_t ws_size,
                              hipStream_t stream) {
  const float* x     = (const float*)d_in[0];
  const float* Wq    = (const float*)d_in[1];
  const float* Wk    = (const float*)d_in[2];
  const float* Wv    = (const float*)d_in[3];
  const float* Wg    = (const float*)d_in[4];
  const float* Wgk1  = (const float*)d_in[5];
  const float* Wgk2  = (const float*)d_in[6];
  const float* bgk2  = (const float*)d_in[7];
  const float* normw = (const float*)d_in[8];
  const float* Wo    = (const float*)d_in[9];

  char* ws = (char*)d_ws;
  size_t off = 0;
  auto alloc = [&](size_t bytes) -> void* {
    void* p = ws + off;
    off = (off + bytes + 255) & ~(size_t)255;
    return p;
  };
  __bf16* xb    = (__bf16*)alloc(MROWS * D_IN * 2);
  __bf16* Wcat  = (__bf16*)alloc((size_t)D_IN * NCAT * 2);
  __bf16* Wob   = (__bf16*)alloc((size_t)D_IN * D_IN * 2);
  __bf16* W1b   = (__bf16*)alloc((size_t)D_IN * 16 * 2);
  float*  projb = (float*)alloc(MROWS * (size_t)NCAT * 4);
  float*  gkb   = (float*)alloc(MROWS * (size_t)DK * 4);
  float*  hb    = (float*)alloc(MROWS * 16 * 4);
  float*  obuf  = (float*)alloc(MROWS * (size_t)DV * 4);
  __bf16* onorm = (__bf16*)alloc(MROWS * (size_t)DV * 2);

  // bf16 conversions
  k_cvt<<<(unsigned)(MROWS * D_IN / 256), 256, 0, stream>>>(x, xb);
  k_wcat<<<(unsigned)((size_t)D_IN * NCAT / 256), 256, 0, stream>>>(Wq, Wk, Wv, Wg, Wcat);
  k_cvt<<<(unsigned)((size_t)D_IN * D_IN / 256), 256, 0, stream>>>(Wo, Wob);
  k_cvt<<<(unsigned)((size_t)D_IN * 16 / 256), 256, 0, stream>>>(Wgk1, W1b);

  // q|k|v|g = x @ [Wq Wk Wv Wg]
  dim3 g1(NCAT / 128, (unsigned)(MROWS / 128));
  k_gemm_bf16<<<g1, 256, 0, stream>>>(xb, Wcat, projb, D_IN, NCAT);

  // low-rank gate path
  k_gemm_h<<<(unsigned)(MROWS / 128), 256, 0, stream>>>(xb, W1b, hb);
  k_gk_act<<<(unsigned)(MROWS * DK / 256), 256, 0, stream>>>(hb, Wgk2, bgk2, gkb);

  // chunked scan: B*H*(dv/64) = 128 blocks
  k_attn<<<128, 256, 0, stream>>>(projb, gkb, obuf);

  // norm + gate, then output projection
  k_norm<<<(unsigned)MROWS, 256, 0, stream>>>(obuf, projb, normw, onorm);
  dim3 g2(D_IN / 128, (unsigned)(MROWS / 128));
  k_gemm_bf16<<<g2, 256, 0, stream>>>(onorm, Wob, (float*)d_out, D_IN, D_IN);
}
